// MoDLayer_41592463294568
// MI455X (gfx1250) — compile-verified, matrix operand order checked by reading, not run on hardware
//
#include <hip/hip_runtime.h>
#include <hip/hip_bf16.h>
#include <math.h>

// ---------------- problem constants (from reference) ----------------
#define DHID  2048
#define NHEAD 16
#define HDIM  128
#define FFD   8192
#define BATCH 2
#define SEQ   4096
#define KSEL  1024          // int(T * 0.25)
#define DQ    512           // D/4
#define RMS_EPS 1e-6f
#define ATTN_SCALE 0.08838834764831845f  // 1/sqrt(128)

typedef __attribute__((ext_vector_type(16))) _Float16 v16h;
typedef __attribute__((ext_vector_type(8)))  _Float16 v8h;
typedef __attribute__((ext_vector_type(8)))  float    v8f;

// ----------------------------------------------------------------------
// WMMA GEMM: C[M,N] = A[M,K] * B[K,N] (+bias[N]); f32 in/out, f16 compute,
// f32 accumulate. Block = 256 threads = 8 waves; block tile 256(M) x 64(N);
// wave tile 32(M) x 64(N) = two A fragments sharing each B fragment.
// Requires M%256==0, N%64==0, K%32==0 (true for all uses here).
//
// CDNA5 16-bit fragment layout (ISA 7.12.2): each lane's 16 halves are two
// contiguous K-runs of 8 (lanes 0-15: K {0..7,16..23}; lanes 16-31:
// K {8..15,24..31}) -> A via 4x global_load_b128 + cvt_pk, B via
// 2x ds_load_b128 from a transposed LDS panel (Bs[n][k], padded stride).
// ----------------------------------------------------------------------
#define BPAD 40   // LDS row stride in halves (80B: 16B-aligned, bank-spread)

__device__ __forceinline__ v16h cvt_afrag(float4 a0, float4 a1,
                                          float4 a2, float4 a3)
{
  v16h f;
  f[0]  = (_Float16)a0.x; f[1]  = (_Float16)a0.y;
  f[2]  = (_Float16)a0.z; f[3]  = (_Float16)a0.w;
  f[4]  = (_Float16)a1.x; f[5]  = (_Float16)a1.y;
  f[6]  = (_Float16)a1.z; f[7]  = (_Float16)a1.w;
  f[8]  = (_Float16)a2.x; f[9]  = (_Float16)a2.y;
  f[10] = (_Float16)a2.z; f[11] = (_Float16)a2.w;
  f[12] = (_Float16)a3.x; f[13] = (_Float16)a3.y;
  f[14] = (_Float16)a3.z; f[15] = (_Float16)a3.w;
  return f;
}

__global__ __launch_bounds__(256) void gemm_f32_wmma(
    const float* __restrict__ A, const float* __restrict__ Bm,
    const float* __restrict__ bias, float* __restrict__ C,
    int M, int N, int K)
{
  __shared__ __align__(16) _Float16 Bs[64 * BPAD];
  const int tid  = threadIdx.x;
  const int lane = tid & 31;
  const int wave = tid >> 5;                 // 0..7
  const int n0   = blockIdx.x * 64;
  const int m0   = blockIdx.y * 256 + wave * 32;
  const int row0 = m0 + (lane & 15);
  const int row1 = row0 + 16;
  const int klo  = (lane < 16) ? 0 : 8;      // first K-run start for this lane

  // staging coordinates (unconditional: 2 chunks x 256 threads = 512 = 32x16)
  const int kk0 = tid >> 4;                  // 0..15
  const int kk1 = kk0 + 16;                  // 16..31
  const int nns = (tid & 15) * 4;            // 0..60

  v8f acc[2][4];
  for (int s = 0; s < 2; ++s)
    for (int f = 0; f < 4; ++f)
      for (int r = 0; r < 8; ++r) acc[s][f][r] = 0.0f;

  const float* Ap0 = A + (size_t)row0 * K + klo;
  const float* Ap1 = A + (size_t)row1 * K + klo;

  for (int kt = 0; kt < K; kt += 32) {
    __syncthreads();
    // stage 32x64 B panel transposed into LDS (Bs[n][k]); both global
    // loads issued before converts so they overlap.
    float4 t0 = *(const float4*)(&Bm[(size_t)(kt + kk0) * N + (n0 + nns)]);
    float4 t1 = *(const float4*)(&Bm[(size_t)(kt + kk1) * N + (n0 + nns)]);
    Bs[(nns + 0) * BPAD + kk0] = (_Float16)t0.x;
    Bs[(nns + 1) * BPAD + kk0] = (_Float16)t0.y;
    Bs[(nns + 2) * BPAD + kk0] = (_Float16)t0.z;
    Bs[(nns + 3) * BPAD + kk0] = (_Float16)t0.w;
    Bs[(nns + 0) * BPAD + kk1] = (_Float16)t1.x;
    Bs[(nns + 1) * BPAD + kk1] = (_Float16)t1.y;
    Bs[(nns + 2) * BPAD + kk1] = (_Float16)t1.z;
    Bs[(nns + 3) * BPAD + kk1] = (_Float16)t1.w;
    __syncthreads();

    // two A fragments (rows row0, row1): 8x global_load_b128 + cvt_pk
    float4 a00 = *(const float4*)(Ap0 + kt);
    float4 a01 = *(const float4*)(Ap0 + kt + 4);
    float4 a02 = *(const float4*)(Ap0 + kt + 16);
    float4 a03 = *(const float4*)(Ap0 + kt + 20);
    float4 a10 = *(const float4*)(Ap1 + kt);
    float4 a11 = *(const float4*)(Ap1 + kt + 4);
    float4 a12 = *(const float4*)(Ap1 + kt + 16);
    float4 a13 = *(const float4*)(Ap1 + kt + 20);
    v16h afrag0 = cvt_afrag(a00, a01, a02, a03);
    v16h afrag1 = cvt_afrag(a10, a11, a12, a13);

#pragma unroll
    for (int f = 0; f < 4; ++f) {
      int col = f * 16 + (lane & 15);
      const _Float16* bp = &Bs[col * BPAD + klo];
      v8h blo = *(const v8h*)(bp);        // ds_load_b128
      v8h bhi = *(const v8h*)(bp + 16);   // ds_load_b128
      v16h bfrag = __builtin_shufflevector(blo, bhi,
          0, 1, 2, 3, 4, 5, 6, 7, 8, 9, 10, 11, 12, 13, 14, 15);
      acc[0][f] = __builtin_amdgcn_wmma_f32_16x16x32_f16(
          false, afrag0, false, bfrag, (short)0, acc[0][f], false, false);
      acc[1][f] = __builtin_amdgcn_wmma_f32_16x16x32_f16(
          false, afrag1, false, bfrag, (short)0, acc[1][f], false, false);
    }
  }

  // C/D layout: VGPR r -> row r (lanes 0-15) / row 8+r (lanes 16-31), col = lane%16
#pragma unroll
  for (int s = 0; s < 2; ++s) {
#pragma unroll
    for (int f = 0; f < 4; ++f) {
      int col = n0 + f * 16 + (lane & 15);
      float bv = bias ? bias[col] : 0.0f;
#pragma unroll
      for (int r = 0; r < 8; ++r) {
        int rr = m0 + s * 16 + r + ((lane < 16) ? 0 : 8);
        C[(size_t)rr * N + col] = acc[s][f][r] + bv;
      }
    }
  }
}

// ---------------- router logits: logits[tok] = dot(hs[tok,:], w) ----------------
__global__ __launch_bounds__(256) void router_kernel(
    const float* __restrict__ hs, const float* __restrict__ w,
    float* __restrict__ logits)
{
  __shared__ float red[256];
  int tok = blockIdx.x;
  const float* in = hs + (size_t)tok * DHID;
  float s = 0.f;
  for (int i = threadIdx.x; i < DHID; i += 256) s += in[i] * w[i];
  red[threadIdx.x] = s; __syncthreads();
  for (int st = 128; st > 0; st >>= 1) {
    if (threadIdx.x < st) red[threadIdx.x] += red[threadIdx.x + st];
    __syncthreads();
  }
  if (threadIdx.x == 0) logits[tok] = red[0];
}

// ---------------- exact top-k: bitonic sort of 4096 keys in LDS ----------------
__global__ __launch_bounds__(1024) void topk_kernel(
    const float* __restrict__ logits, int* __restrict__ topk_idx,
    float* __restrict__ g, float* __restrict__ sel)
{
  __shared__ unsigned long long keys[SEQ];
  int b = blockIdx.x;
  const float* lrow = logits + (size_t)b * SEQ;
  for (int i = threadIdx.x; i < SEQ; i += 1024) {
    unsigned u = __float_as_uint(lrow[i]);
    u = (u & 0x80000000u) ? ~u : (u | 0x80000000u);   // order-preserving
    keys[i] = ((unsigned long long)u << 32) | (unsigned)(SEQ - 1 - i);
  }
  __syncthreads();
  for (int size = 2; size <= SEQ; size <<= 1) {
    for (int stride = size >> 1; stride > 0; stride >>= 1) {
      for (int i = threadIdx.x; i < SEQ; i += 1024) {
        int jj = i ^ stride;
        if (jj > i) {
          bool desc = ((i & size) == 0);
          unsigned long long a = keys[i], c = keys[jj];
          bool sw = desc ? (a < c) : (a > c);
          if (sw) { keys[i] = c; keys[jj] = a; }
        }
      }
      __syncthreads();
    }
  }
  if (threadIdx.x < KSEL) {
    unsigned long long kkey = keys[threadIdx.x];
    int src = SEQ - 1 - (int)(kkey & 0xFFFFFFFFu);
    topk_idx[b * KSEL + threadIdx.x] = src;
    float lg = lrow[src];
    g[b * KSEL + threadIdx.x] = 1.0f / (1.0f + __expf(-lg));
    sel[(size_t)b * SEQ + src] = 1.0f;
  }
}

// -------- pred_logits[row] = sum_f gelu(h[row,f]) * fc2_w[f] + fc2_b --------
__global__ __launch_bounds__(256) void predlogit_kernel(
    const float* __restrict__ h, const float* __restrict__ w2,
    const float* __restrict__ b2, float* __restrict__ out)
{
  __shared__ float red[256];
  int rowi = blockIdx.x;
  const float* hr = h + (size_t)rowi * DQ;
  float s = 0.f;
  for (int i = threadIdx.x; i < DQ; i += 256) {
    float x = hr[i];
    float ge = 0.5f * x * (1.0f + erff(x * 0.70710678118654752f));
    s += ge * w2[i];
  }
  red[threadIdx.x] = s; __syncthreads();
  for (int st = 128; st > 0; st >>= 1) {
    if (threadIdx.x < st) red[threadIdx.x] += red[threadIdx.x + st];
    __syncthreads();
  }
  if (threadIdx.x == 0) out[rowi] = red[0] + b2[0];
}

// ---------------- losses (BCE router, z, BCE predictor) ----------------
__global__ __launch_bounds__(256) void loss_kernel(
    const float* __restrict__ logits, const float* __restrict__ pred,
    const float* __restrict__ sel, float* __restrict__ accum, int n)
{
  __shared__ float r0[256], r1[256], r2[256];
  float s_bce = 0.f, s_z = 0.f, s_p = 0.f;
  for (int i = blockIdx.x * 256 + threadIdx.x; i < n; i += gridDim.x * 256) {
    float x = logits[i], y = sel[i];
    s_bce += fmaxf(x, 0.f) - x * y + log1pf(__expf(-fabsf(x)));
    s_z   += x * x;
    float xp = pred[i];
    s_p   += fmaxf(xp, 0.f) - xp * y + log1pf(__expf(-fabsf(xp)));
  }
  r0[threadIdx.x] = s_bce; r1[threadIdx.x] = s_z; r2[threadIdx.x] = s_p;
  __syncthreads();
  for (int st = 128; st > 0; st >>= 1) {
    if (threadIdx.x < st) {
      r0[threadIdx.x] += r0[threadIdx.x + st];
      r1[threadIdx.x] += r1[threadIdx.x + st];
      r2[threadIdx.x] += r2[threadIdx.x + st];
    }
    __syncthreads();
  }
  if (threadIdx.x == 0) {
    atomicAdd(&accum[0], r0[0]);
    atomicAdd(&accum[1], r1[0]);
    atomicAdd(&accum[2], r2[0]);
  }
}

__global__ void finalize_losses(const float* __restrict__ accum,
                                float* __restrict__ out3, float invn)
{
  if (blockIdx.x == 0 && threadIdx.x == 0) {
    out3[0] = accum[0] * invn;           // router_bce
    out3[1] = accum[1] * invn * 1e-4f;   // z loss
    out3[2] = accum[2] * invn;           // predictor bce
  }
}

// ---------------- gather selected tokens + RMSNorm(ln1) ----------------
__global__ __launch_bounds__(256) void gather_rmsnorm_kernel(
    const float* __restrict__ hs, const int* __restrict__ idx,
    const float* __restrict__ lnw, float* __restrict__ x_sel,
    float* __restrict__ xa)
{
  __shared__ float red[256];
  int tok = blockIdx.x;                 // 0..B*KSEL-1
  int b = tok / KSEL;
  int src = idx[tok];
  const float* in = hs + ((size_t)b * SEQ + src) * DHID;
  float* xs = x_sel + (size_t)tok * DHID;
  float* xo = xa    + (size_t)tok * DHID;
  float loc[8]; float ss = 0.f;
#pragma unroll
  for (int i = 0; i < 8; ++i) {
    float v = in[threadIdx.x + i * 256];
    loc[i] = v; ss += v * v;
  }
  red[threadIdx.x] = ss; __syncthreads();
  for (int st = 128; st > 0; st >>= 1) {
    if (threadIdx.x < st) red[threadIdx.x] += red[threadIdx.x + st];
    __syncthreads();
  }
  float rinv = rsqrtf(red[0] / (float)DHID + RMS_EPS);
#pragma unroll
  for (int i = 0; i < 8; ++i) {
    int c = threadIdx.x + i * 256;
    xs[c] = loc[i];
    xo[c] = loc[i] * rinv * lnw[c];
  }
}

// ---------------- causal attention, online softmax, 1 wave / (b,h,row) ----------------
__global__ __launch_bounds__(256) void attn_kernel(
    const float* __restrict__ Q, const float* __restrict__ Km,
    const float* __restrict__ V, float* __restrict__ O)
{
  int gid  = blockIdx.x * 8 + (threadIdx.x >> 5);  // wave id, total B*H*KSEL
  int lane = threadIdx.x & 31;
  int b  = gid / (NHEAD * KSEL);
  int r  = gid % (NHEAD * KSEL);
  int h  = r / KSEL;
  int qi = r % KSEL;

  size_t qbase = ((size_t)(b * KSEL + qi)) * DHID + h * HDIM + lane * 4;
  float4 qv = *(const float4*)(Q + qbase);

  float m = -3.0e38f, ssum = 0.f;
  float4 acc; acc.x = acc.y = acc.z = acc.w = 0.f;

  for (int j = 0; j <= qi; ++j) {
    size_t kb = ((size_t)(b * KSEL + j)) * DHID + h * HDIM + lane * 4;
    float4 kv = *(const float4*)(Km + kb);
    float part = qv.x * kv.x + qv.y * kv.y + qv.z * kv.z + qv.w * kv.w;
    for (int msk = 16; msk > 0; msk >>= 1)
      part += __shfl_xor(part, msk, 32);
    float s = part * ATTN_SCALE;
    float mnew = fmaxf(m, s);
    float corr = __expf(m - mnew);
    float p    = __expf(s - mnew);
    ssum = ssum * corr + p;
    float4 vv = *(const float4*)(V + kb);
    acc.x = acc.x * corr + p * vv.x;
    acc.y = acc.y * corr + p * vv.y;
    acc.z = acc.z * corr + p * vv.z;
    acc.w = acc.w * corr + p * vv.w;
    m = mnew;
  }
  float inv = 1.0f / ssum;
  float4 res; res.x = acc.x * inv; res.y = acc.y * inv;
  res.z = acc.z * inv; res.w = acc.w * inv;
  *(float4*)(O + qbase) = res;
}

// ---------------- x2 = x_sel + attn_proj ; xm = RMSNorm(x2)*ln2 ----------------
__global__ __launch_bounds__(256) void resid_rmsnorm_kernel(
    const float* __restrict__ x_sel, const float* __restrict__ ao,
    const float* __restrict__ lnw, float* __restrict__ x2,
    float* __restrict__ xm)
{
  __shared__ float red[256];
  int tok = blockIdx.x;
  const float* a  = x_sel + (size_t)tok * DHID;
  const float* bo = ao    + (size_t)tok * DHID;
  float loc[8]; float ss = 0.f;
#pragma unroll
  for (int i = 0; i < 8; ++i) {
    int c = threadIdx.x + i * 256;
    float v = a[c] + bo[c];
    loc[i] = v; ss += v * v;
  }
  red[threadIdx.x] = ss; __syncthreads();
  for (int st = 128; st > 0; st >>= 1) {
    if (threadIdx.x < st) red[threadIdx.x] += red[threadIdx.x + st];
    __syncthreads();
  }
  float rinv = rsqrtf(red[0] / (float)DHID + RMS_EPS);
#pragma unroll
  for (int i = 0; i < 8; ++i) {
    int c = threadIdx.x + i * 256;
    x2[(size_t)tok * DHID + c] = loc[i];
    xm[(size_t)tok * DHID + c] = loc[i] * rinv * lnw[c];
  }
}

// ---------------- act = silu(gate) * up (in place into gate) ----------------
__global__ void silu_mul_kernel(float* __restrict__ gate,
                                const float* __restrict__ up, size_t n)
{
  size_t i = (size_t)blockIdx.x * blockDim.x + threadIdx.x;
  size_t stride = (size_t)gridDim.x * blockDim.x;
  for (; i < n; i += stride) {
    float gv = gate[i];
    gate[i] = (gv / (1.0f + __expf(-gv))) * up[i];
  }
}

// ---------------- out = hidden_states (full copy) ----------------
__global__ void copy_kernel(const float* __restrict__ in,
                            float* __restrict__ out, size_t n)
{
  size_t i = (size_t)blockIdx.x * blockDim.x + threadIdx.x;
  size_t stride = (size_t)gridDim.x * blockDim.x;
  for (; i < n; i += stride) out[i] = in[i];
}

// ---------------- scatter gated block output back into out ----------------
__global__ __launch_bounds__(256) void scatter_kernel(
    const float* __restrict__ x_sel, const float* __restrict__ x2,
    const float* __restrict__ down, const float* __restrict__ g,
    const int* __restrict__ idx, float* __restrict__ out)
{
  int tok = blockIdx.x;
  int b = tok / KSEL;
  int src = idx[tok];
  float gg = g[tok];
  float* o = out + ((size_t)b * SEQ + src) * DHID;
  const float* xs = x_sel + (size_t)tok * DHID;
  const float* a2 = x2    + (size_t)tok * DHID;
  const float* dn = down  + (size_t)tok * DHID;
  for (int c = threadIdx.x; c < DHID; c += 256) {
    float bo = a2[c] + dn[c];                  // block_out
    o[c] = xs[c] + gg * (bo - xs[c]);
  }
}

// ---------------- zero sel mask + loss accumulators ----------------
__global__ void init_kernel(float* __restrict__ sel, float* __restrict__ accum, int n)
{
  int i = blockIdx.x * blockDim.x + threadIdx.x;
  if (i < n) sel[i] = 0.f;
  if (i < 3) accum[i] = 0.f;
}

// ======================================================================
extern "C" void kernel_launch(void* const* d_in, const int* in_sizes, int n_in,
                              void* d_out, int out_size, void* d_ws, size_t ws_size,
                              hipStream_t stream)
{
  (void)in_sizes; (void)n_in; (void)out_size; (void)ws_size;
  const float* hs       = (const float*)d_in[0];
  const float* router_w = (const float*)d_in[1];
  const float* fc1_w    = (const float*)d_in[2];
  const float* fc1_b    = (const float*)d_in[3];
  const float* fc2_w    = (const float*)d_in[4];
  const float* fc2_b    = (const float*)d_in[5];
  const float* ln1_w    = (const float*)d_in[6];
  const float* ln2_w    = (const float*)d_in[7];
  const float* wq       = (const float*)d_in[8];
  const float* bq       = (const float*)d_in[9];
  const float* wk       = (const float*)d_in[10];
  const float* bk       = (const float*)d_in[11];
  const float* wv       = (const float*)d_in[12];
  const float* bv       = (const float*)d_in[13];
  const float* wo       = (const float*)d_in[14];
  const float* w_gate   = (const float*)d_in[15];
  const float* w_up     = (const float*)d_in[16];
  const float* w_down   = (const float*)d_in[17];

  float* out = (float*)d_out;                       // [B,T,D] then 3 losses
  float* out_losses = out + (size_t)BATCH * SEQ * DHID;

  // ---- workspace arena (floats); total ~224 MiB ----
  float* ws = (float*)d_ws;
  const size_t NTOK  = (size_t)BATCH * SEQ;         // 8192
  const size_t NSELR = (size_t)BATCH * KSEL;        // 2048 rows
  const size_t ROWSZ = NSELR * DHID;                // 4,194,304
  float* logits  = ws;                              // 8192
  float* predl   = ws + 8192;                       // 8192
  float* sel     = ws + 16384;                      // 8192
  float* accum   = ws + 24576;                      // 3 (+pad)
  float* gbuf    = ws + 24608;                      // 2048
  int*   topkidx = (int*)(ws + 26656);              // 2048
  float* predh   = ws + 32768;                      // 8192*512 ; reused: mlp-down out
  float* x_sel   = predh + (size_t)NTOK * DQ;       // ROWSZ
  float* xa      = x_sel + ROWSZ;                   // ROWSZ ; reused: attn out
  float* qbuf    = xa + ROWSZ;                      // ROWSZ ; reused: o-proj out
  float* kbuf    = qbuf + ROWSZ;                    // ROWSZ ; reused: x2
  float* vbuf    = kbuf + ROWSZ;                    // ROWSZ ; reused: xm
  float* gatebuf = vbuf + ROWSZ;                    // NSELR*FFD
  float* upbuf   = gatebuf + NSELR * (size_t)FFD;   // NSELR*FFD

  // 0) init masks / accumulators
  init_kernel<<<(NTOK + 255) / 256, 256, 0, stream>>>(sel, accum, (int)NTOK);

  // 1) router logits
  router_kernel<<<NTOK, 256, 0, stream>>>(hs, router_w, logits);

  // 2) predictor fc1 GEMM: [8192,2048] x [2048,512] (+b)  -> predh
  gemm_f32_wmma<<<dim3(DQ / 64, NTOK / 256), 256, 0, stream>>>(
      hs, fc1_w, fc1_b, predh, (int)NTOK, DQ, DHID);

  // 3) exact top-k per batch row (bitonic sort in LDS)
  topk_kernel<<<BATCH, 1024, 0, stream>>>(logits, topkidx, gbuf, sel);

  // 4) predictor logits: gelu + dot with fc2_w
  predlogit_kernel<<<NTOK, 256, 0, stream>>>(predh, fc2_w, fc2_b, predl);

  // 5) losses
  loss_kernel<<<64, 256, 0, stream>>>(logits, predl, sel, accum, (int)NTOK);
  finalize_losses<<<1, 32, 0, stream>>>(accum, out_losses, 1.0f / (float)NTOK);

  // 6) gather + RMSNorm(ln1)
  gather_rmsnorm_kernel<<<NSELR, 256, 0, stream>>>(hs, topkidx, ln1_w, x_sel, xa);

  // 7) QKV projections (WMMA)
  gemm_f32_wmma<<<dim3(DHID / 64, NSELR / 256), 256, 0, stream>>>(
      xa, wq, bq, qbuf, (int)NSELR, DHID, DHID);
  gemm_f32_wmma<<<dim3(DHID / 64, NSELR / 256), 256, 0, stream>>>(
      xa, wk, bk, kbuf, (int)NSELR, DHID, DHID);
  gemm_f32_wmma<<<dim3(DHID / 64, NSELR / 256), 256, 0, stream>>>(
      xa, wv, bv, vbuf, (int)NSELR, DHID, DHID);

  // 8) causal attention (online softmax), writes into xa
  attn_kernel<<<(BATCH * NHEAD * KSEL) / 8, 256, 0, stream>>>(qbuf, kbuf, vbuf, xa);

  // 9) output projection -> qbuf (no bias)
  gemm_f32_wmma<<<dim3(DHID / 64, NSELR / 256), 256, 0, stream>>>(
      xa, wo, nullptr, qbuf, (int)NSELR, DHID, DHID);

  // 10) x2 = x_sel + o ; xm = RMSNorm(x2)*ln2  (x2->kbuf, xm->vbuf)
  resid_rmsnorm_kernel<<<NSELR, 256, 0, stream>>>(x_sel, qbuf, ln2_w, kbuf, vbuf);

  // 11) MLP gate / up GEMMs (WMMA)
  gemm_f32_wmma<<<dim3(FFD / 64, NSELR / 256), 256, 0, stream>>>(
      vbuf, w_gate, nullptr, gatebuf, (int)NSELR, FFD, DHID);
  gemm_f32_wmma<<<dim3(FFD / 64, NSELR / 256), 256, 0, stream>>>(
      vbuf, w_up, nullptr, upbuf, (int)NSELR, FFD, DHID);

  // 12) act = silu(gate)*up
  silu_mul_kernel<<<4096, 256, 0, stream>>>(gatebuf, upbuf, NSELR * (size_t)FFD);

  // 13) down GEMM: [2048,8192] x [8192,2048] -> predh (reused)
  gemm_f32_wmma<<<dim3(DHID / 64, NSELR / 256), 256, 0, stream>>>(
      gatebuf, w_down, nullptr, predh, (int)NSELR, DHID, FFD);

  // 14) out = hidden_states ; then scatter gated block output
  copy_kernel<<<4096, 256, 0, stream>>>(hs, out, (size_t)NTOK * DHID);
  scatter_kernel<<<NSELR, 256, 0, stream>>>(x_sel, kbuf, predh, gbuf, topkidx, out);
}
